// RobustFill_59820304499080
// MI455X (gfx1250) — matrix operand order repaired and checked by reading, not compiled, not run on hardware
//
#include <hip/hip_runtime.h>
#include <hip/hip_bf16.h>
#include <math.h>

// Problem constants (from reference): S=128, H=1024, P=1024, NL=2, PL=32, T_IN=T_OUT=64
#define H_     1024
#define G_     4096   /* 4*H gate rows */
#define S_     128
#define T_IN_  64
#define T_OUT_ 64
#define PL_    32
#define P_     1024

typedef __attribute__((ext_vector_type(16))) __bf16          bf16x16;
typedef __attribute__((ext_vector_type(8)))  float           f32x8;
typedef __attribute__((ext_vector_type(8)))  unsigned short  u16x8;
typedef __attribute__((ext_vector_type(16))) unsigned short  u16x16;

// float -> bf16 (round to nearest even)
static __device__ __forceinline__ unsigned short f2bf(float f) {
  unsigned u = __float_as_uint(f);
  unsigned r = u + 0x7fffu + ((u >> 16) & 1u);
  return (unsigned short)(r >> 16);
}

// A fragment (16x32 bf16, M x K) from row-major bf16 matrix, tileBase = first row of tile.
// Contiguous k order used here: lanes 0-15: M=lane, K=0..15 ; lanes 16-31: M=lane-16, K=16..31.
static __device__ __forceinline__ bf16x16 load_a_frag(const unsigned short* __restrict__ tileBase,
                                                      int ld, int k0, int lane) {
  const unsigned short* p = tileBase + (size_t)(lane & 15) * ld + k0 + ((lane >> 4) << 4);
  u16x8 lo = *reinterpret_cast<const u16x8*>(p);
  u16x8 hi = *reinterpret_cast<const u16x8*>(p + 8);
  u16x16 v;
#pragma unroll
  for (int i = 0; i < 8; i++) { v[i] = lo[i]; v[8 + i] = hi[i]; }
  return __builtin_bit_cast(bf16x16, v);
}

// B fragment (32x16 bf16, K x N) where column n = row n of a row-major bf16 matrix (i.e. B = M^T).
// Contiguous k order (same as A): lanes 0-15 hold K=0..15 of column lane; lanes 16-31 hold K=16..31 of column lane-16.
static __device__ __forceinline__ bf16x16 load_b_frag(const unsigned short* __restrict__ tileBase,
                                                      int ld, int k0, int lane) {
  const unsigned short* p = tileBase + (size_t)(lane & 15) * ld + k0 + ((lane >> 4) << 4);
  u16x8 lo = *reinterpret_cast<const u16x8*>(p);
  u16x8 hi = *reinterpret_cast<const u16x8*>(p + 8);
  u16x16 v;
#pragma unroll
  for (int i = 0; i < 8; i++) { v[i] = lo[i]; v[8 + i] = hi[i]; }
  return __builtin_bit_cast(bf16x16, v);
}

// GEMV B fragment: broadcast the same bf16 vector chunk into ALL 16 columns.
// No masking needed — D[m,n] depends only on column n of B, so every column of D
// computes the identical dot product; we read column 0 and ignore the rest.
static __device__ __forceinline__ bf16x16 vec_b_frag(const unsigned short* __restrict__ vb,
                                                     int k0, int lane) {
  const unsigned short* p = vb + k0 + ((lane >> 4) << 4);   // lanes 0-15: K 0..15 ; 16-31: K 16..31
  u16x8 lo = *reinterpret_cast<const u16x8*>(p);
  u16x8 hi = *reinterpret_cast<const u16x8*>(p + 8);
  u16x16 v;
#pragma unroll
  for (int i = 0; i < 8; i++) { v[i] = lo[i]; v[8 + i] = hi[i]; }
  return __builtin_bit_cast(bf16x16, v);
}

// ---------------------------------------------------------------------------
// gates[r] = (base[r]  OR  b1[r]+b2[r])  (+ onehotW[r, seq[t]])
//            + Whh[r,:] @ hb  (+ Wih[r,:] @ xb)       (hb/xb are bf16 vectors)
// One wave per block, 16 gate rows per block, 256 blocks -> 4096 rows.
// Dual accumulators break the WMMA->WMMA RAW chain (no hazard NOPs).
// ---------------------------------------------------------------------------
__global__ __launch_bounds__(32) void lstm_gates_kernel(
    const unsigned short* __restrict__ Whh, const unsigned short* __restrict__ hb,
    const unsigned short* __restrict__ Wih, const unsigned short* __restrict__ xb,
    const float* __restrict__ base,
    const float* __restrict__ b1, const float* __restrict__ b2,
    const float* __restrict__ onehotW, int onehot_ld,
    const int* __restrict__ seq, int t,
    float* __restrict__ gates)
{
  const int lane = threadIdx.x;
  const int r0 = blockIdx.x * 16;

  f32x8 acc0 = {}, acc1 = {};
  const unsigned short* Wt = Whh + (size_t)r0 * H_;
#pragma unroll 2
  for (int k = 0; k < H_; k += 64) {
    __builtin_prefetch(Wt + (size_t)(lane & 15) * H_ + k + 128, 0, 3);
    bf16x16 a0 = load_a_frag(Wt, H_, k,      lane);
    bf16x16 b0 = vec_b_frag(hb, k,      lane);
    bf16x16 a1 = load_a_frag(Wt, H_, k + 32, lane);
    bf16x16 b1v = vec_b_frag(hb, k + 32, lane);
    acc0 = __builtin_amdgcn_wmma_f32_16x16x32_bf16(false, a0, false, b0,  (short)0, acc0, false, false);
    acc1 = __builtin_amdgcn_wmma_f32_16x16x32_bf16(false, a1, false, b1v, (short)0, acc1, false, false);
  }
  if (Wih != nullptr) {
    const unsigned short* Wt2 = Wih + (size_t)r0 * H_;
#pragma unroll 2
    for (int k = 0; k < H_; k += 64) {
      __builtin_prefetch(Wt2 + (size_t)(lane & 15) * H_ + k + 128, 0, 3);
      bf16x16 a0 = load_a_frag(Wt2, H_, k,      lane);
      bf16x16 b0 = vec_b_frag(xb, k,      lane);
      bf16x16 a1 = load_a_frag(Wt2, H_, k + 32, lane);
      bf16x16 b1v = vec_b_frag(xb, k + 32, lane);
      acc0 = __builtin_amdgcn_wmma_f32_16x16x32_bf16(false, a0, false, b0,  (short)0, acc0, false, false);
      acc1 = __builtin_amdgcn_wmma_f32_16x16x32_bf16(false, a1, false, b1v, (short)0, acc1, false, false);
    }
  }
  f32x8 acc = acc0 + acc1;

  // Every column of D holds the same GEMV result; lane 0 (M=0..7) and lane 16 (M=8..15) store.
  if ((lane & 15) == 0) {
    const int m0 = r0 + ((lane >> 4) << 3);
    const int idx = seq ? seq[t] : 0;
#pragma unroll
    for (int i = 0; i < 8; i++) {
      const int row = m0 + i;
      float g = acc[i];
      g += base ? base[row] : (b1[row] + b2[row]);
      if (onehotW) g += onehotW[(size_t)row * onehot_ld + idx];
      gates[row] = g;
    }
  }
}

// LSTM cell pointwise update; PyTorch gate order i,f,g,o in gates[0..4H).
// c stays f32 (accumulating recurrence); h is emitted as bf16 (all consumers are WMMA operands).
__global__ void lstm_cell_kernel(const float* __restrict__ gates,
                                 unsigned short* __restrict__ hb, float* __restrict__ c,
                                 unsigned short* __restrict__ ys_row)
{
  int j = blockIdx.x * blockDim.x + threadIdx.x;
  if (j < H_) {
    float ig = 1.f / (1.f + expf(-gates[j]));
    float fg = 1.f / (1.f + expf(-gates[H_ + j]));
    float gg = tanhf(gates[2 * H_ + j]);
    float og = 1.f / (1.f + expf(-gates[3 * H_ + j]));
    float cn = fg * c[j] + ig * gg;
    float hn = og * tanhf(cn);
    c[j] = cn;
    unsigned short hv = f2bf(hn);
    hb[j] = hv;
    if (ys_row) ys_row[j] = hv;
  }
}

// Dense head GEMM: out[m,n] = ys[m,:] . W[n,:] + bias[n], M=32, N=K=1024. Full WMMA tiles.
__global__ __launch_bounds__(32) void head_gemm_kernel(
    const unsigned short* __restrict__ ysb,   // [PL_, H_] bf16
    const unsigned short* __restrict__ Wb,    // [P_,  H_] bf16 (lin_W row-major)
    const float* __restrict__ bias,           // [P_]
    float* __restrict__ out)                  // [PL_, P_] f32
{
  const int lane  = threadIdx.x;
  const int ntile = blockIdx.x;   // 0..63
  const int mtile = blockIdx.y;   // 0..1
  f32x8 acc0 = {}, acc1 = {};
  const unsigned short* A = ysb + (size_t)(mtile * 16) * H_;
  const unsigned short* B = Wb  + (size_t)(ntile * 16) * H_;
#pragma unroll 2
  for (int k = 0; k < H_; k += 64) {
    bf16x16 a0 = load_a_frag(A, H_, k,      lane);
    bf16x16 b0 = load_b_frag(B, H_, k,      lane);   // column n of B = row n of lin_W
    bf16x16 a1 = load_a_frag(A, H_, k + 32, lane);
    bf16x16 b1v = load_b_frag(B, H_, k + 32, lane);
    acc0 = __builtin_amdgcn_wmma_f32_16x16x32_bf16(false, a0, false, b0,  (short)0, acc0, false, false);
    acc1 = __builtin_amdgcn_wmma_f32_16x16x32_bf16(false, a1, false, b1v, (short)0, acc1, false, false);
  }
  f32x8 acc = acc0 + acc1;
  const int n  = ntile * 16 + (lane & 15);
  const int m0 = mtile * 16 + ((lane >> 4) << 3);
  const float bb = bias[n];
#pragma unroll
  for (int i = 0; i < 8; i++)
    out[(size_t)(m0 + i) * P_ + n] = acc[i] + bb;
}

__global__ void f2bf_kernel(const float* __restrict__ src, unsigned short* __restrict__ dst, int n) {
  int i = blockIdx.x * blockDim.x + threadIdx.x;
  if (i < n) dst[i] = f2bf(src[i]);
}

__global__ void zero_f32_kernel(float* __restrict__ p, int n) {
  int i = blockIdx.x * blockDim.x + threadIdx.x;
  if (i < n) p[i] = 0.f;
}

__global__ void zero_u16_kernel(unsigned short* __restrict__ p, int n) {
  int i = blockIdx.x * blockDim.x + threadIdx.x;
  if (i < n) p[i] = 0;
}

// ---------------------------------------------------------------------------
extern "C" void kernel_launch(void* const* d_in, const int* in_sizes, int n_in,
                              void* d_out, int out_size, void* d_ws, size_t ws_size,
                              hipStream_t stream)
{
  (void)in_sizes; (void)n_in; (void)out_size; (void)ws_size;

  const float* in_Wih0  = (const float*)d_in[0];
  const float* in_Whh0  = (const float*)d_in[1];
  const float* in_bih0  = (const float*)d_in[2];
  const float* in_bhh0  = (const float*)d_in[3];
  const float* in_Wih1  = (const float*)d_in[4];
  const float* in_Whh1  = (const float*)d_in[5];
  const float* in_bih1  = (const float*)d_in[6];
  const float* in_bhh1  = (const float*)d_in[7];
  const float* out_Wih0 = (const float*)d_in[8];
  const float* out_Whh0 = (const float*)d_in[9];
  const float* out_bih0 = (const float*)d_in[10];
  const float* out_bhh0 = (const float*)d_in[11];
  const float* out_Wih1 = (const float*)d_in[12];
  const float* out_Whh1 = (const float*)d_in[13];
  const float* out_bih1 = (const float*)d_in[14];
  const float* out_bhh1 = (const float*)d_in[15];
  const float* pr_Wih0  = (const float*)d_in[16];
  const float* pr_Whh0  = (const float*)d_in[17];
  const float* pr_bih0  = (const float*)d_in[18];
  const float* pr_bhh0  = (const float*)d_in[19];
  const float* pr_Wih1  = (const float*)d_in[20];
  const float* pr_Whh1  = (const float*)d_in[21];
  const float* pr_bih1  = (const float*)d_in[22];
  const float* pr_bhh1  = (const float*)d_in[23];
  const float* lin_W    = (const float*)d_in[24];
  const float* lin_b    = (const float*)d_in[25];
  const int*   input_seq  = (const int*)d_in[26];
  const int*   output_seq = (const int*)d_in[27];
  float* out = (float*)d_out;

  // ---- workspace carve-out (256B aligned) ----
  char* ws = (char*)d_ws;
  size_t off = 0;
  auto alloc = [&](size_t bytes) -> char* {
    char* p = ws + off;
    off = (off + bytes + 255) & ~(size_t)255;
    return p;
  };
  const size_t WB = (size_t)G_ * H_ * sizeof(unsigned short);   // 8 MB per bf16 weight matrix
  unsigned short* bWhh_in0  = (unsigned short*)alloc(WB);
  unsigned short* bWih_in1  = (unsigned short*)alloc(WB);
  unsigned short* bWhh_in1  = (unsigned short*)alloc(WB);
  unsigned short* bWhh_out0 = (unsigned short*)alloc(WB);
  unsigned short* bWih_out1 = (unsigned short*)alloc(WB);
  unsigned short* bWhh_out1 = (unsigned short*)alloc(WB);
  unsigned short* bWih_pr0  = (unsigned short*)alloc(WB);
  unsigned short* bWhh_pr0  = (unsigned short*)alloc(WB);
  unsigned short* bWih_pr1  = (unsigned short*)alloc(WB);
  unsigned short* bWhh_pr1  = (unsigned short*)alloc(WB);
  unsigned short* bLinW     = (unsigned short*)alloc((size_t)P_ * H_ * sizeof(unsigned short));
  unsigned short* h0b = (unsigned short*)alloc(H_ * sizeof(unsigned short));
  unsigned short* h1b = (unsigned short*)alloc(H_ * sizeof(unsigned short));
  float* cstate   = (float*)alloc(2 * H_ * sizeof(float));      // c0,c1 contiguous
  float* c0 = cstate, *c1 = cstate + H_;
  float* gates    = (float*)alloc((size_t)G_ * sizeof(float));
  float* pr0_base = (float*)alloc((size_t)G_ * sizeof(float));
  unsigned short* ysb = (unsigned short*)alloc((size_t)PL_ * H_ * sizeof(unsigned short));

  // ---- one-time per call: bf16 weight conversion (memory-bound GEMVs -> half the bytes) ----
  auto conv = [&](const float* s, unsigned short* d, int n) {
    f2bf_kernel<<<dim3((n + 255) / 256), dim3(256), 0, stream>>>(s, d, n);
  };
  conv(in_Whh0,  bWhh_in0,  G_ * H_);
  conv(in_Wih1,  bWih_in1,  G_ * H_);
  conv(in_Whh1,  bWhh_in1,  G_ * H_);
  conv(out_Whh0, bWhh_out0, G_ * H_);
  conv(out_Wih1, bWih_out1, G_ * H_);
  conv(out_Whh1, bWhh_out1, G_ * H_);
  conv(pr_Wih0,  bWih_pr0,  G_ * H_);
  conv(pr_Whh0,  bWhh_pr0,  G_ * H_);
  conv(pr_Wih1,  bWih_pr1,  G_ * H_);
  conv(pr_Whh1,  bWhh_pr1,  G_ * H_);
  conv(lin_W,    bLinW,     P_ * H_);

  zero_f32_kernel<<<dim3((2 * H_ + 255) / 256), dim3(256), 0, stream>>>(cstate, 2 * H_);
  zero_u16_kernel<<<dim3((H_ + 255) / 256), dim3(256), 0, stream>>>(h0b, H_);
  zero_u16_kernel<<<dim3((H_ + 255) / 256), dim3(256), 0, stream>>>(h1b, H_);

  const dim3 gG(256), gB(32);        // gates: 256 tiles x 1 wave
  const dim3 cG(4),  cB(256);        // cell:  1024 lanes

  // ---- input encoder: layer0 Wih@one_hot is a column gather ----
  for (int t = 0; t < T_IN_; ++t) {
    lstm_gates_kernel<<<gG, gB, 0, stream>>>(bWhh_in0, h0b, nullptr, nullptr,
        nullptr, in_bih0, in_bhh0, in_Wih0, S_, input_seq, t, gates);
    lstm_cell_kernel<<<cG, cB, 0, stream>>>(gates, h0b, c0, nullptr);
    lstm_gates_kernel<<<gG, gB, 0, stream>>>(bWhh_in1, h1b, bWih_in1, h0b,
        nullptr, in_bih1, in_bhh1, nullptr, 0, nullptr, 0, gates);
    lstm_cell_kernel<<<cG, cB, 0, stream>>>(gates, h1b, c1, nullptr);
  }

  // ---- output encoder ----
  for (int t = 0; t < T_OUT_; ++t) {
    lstm_gates_kernel<<<gG, gB, 0, stream>>>(bWhh_out0, h0b, nullptr, nullptr,
        nullptr, out_bih0, out_bhh0, out_Wih0, S_, output_seq, t, gates);
    lstm_cell_kernel<<<cG, cB, 0, stream>>>(gates, h0b, c0, nullptr);
    lstm_gates_kernel<<<gG, gB, 0, stream>>>(bWhh_out1, h1b, bWih_out1, h0b,
        nullptr, out_bih1, out_bhh1, nullptr, 0, nullptr, 0, gates);
    lstm_cell_kernel<<<cG, cB, 0, stream>>>(gates, h1b, c1, nullptr);
  }

  // ---- program LSTM: constant input => hoist pr_Wih0 @ prev_h (+biases) out of the loop ----
  lstm_gates_kernel<<<gG, gB, 0, stream>>>(bWih_pr0, h1b, nullptr, nullptr,
      nullptr, pr_bih0, pr_bhh0, nullptr, 0, nullptr, 0, pr0_base);
  for (int t = 0; t < PL_; ++t) {
    lstm_gates_kernel<<<gG, gB, 0, stream>>>(bWhh_pr0, h0b, nullptr, nullptr,
        pr0_base, nullptr, nullptr, nullptr, 0, nullptr, 0, gates);
    lstm_cell_kernel<<<cG, cB, 0, stream>>>(gates, h0b, c0, nullptr);
    lstm_gates_kernel<<<gG, gB, 0, stream>>>(bWhh_pr1, h1b, bWih_pr1, h0b,
        nullptr, pr_bih1, pr_bhh1, nullptr, 0, nullptr, 0, gates);
    lstm_cell_kernel<<<cG, cB, 0, stream>>>(gates, h1b, c1, ysb + (size_t)t * H_);
  }

  // ---- dense head: [32,1024] @ [1024,1024]^T with full-tile WMMA ----
  head_gemm_kernel<<<dim3(P_ / 16, PL_ / 16), dim3(32), 0, stream>>>(ysb, bLinW, lin_b, out);
}
